// EdgeDyFraudNet_44117904065165
// MI455X (gfx1250) — matrix operand order repaired and checked by reading, not compiled, not run on hardware
//
#include <hip/hip_runtime.h>
#include <hip/hip_bf16.h>

#define NNODES 50000
#define NEDGES 800000
#define NLBL   400000
#define HDIM   64

typedef __attribute__((ext_vector_type(16))) __bf16          v16bf;
typedef __attribute__((ext_vector_type(8)))  float           v8f;
typedef __attribute__((ext_vector_type(8)))  unsigned short  v8us;

__device__ __forceinline__ unsigned short f2bf(float f) {
    unsigned u = __builtin_bit_cast(unsigned, f);
    unsigned r = u + 0x7FFFu + ((u >> 16) & 1u);   // round-to-nearest-even
    return (unsigned short)(r >> 16);
}

struct bfpair { v8us lo, hi; };

// A 16x32 bf16 fragment: lane holds row lr; K = kb+hf*8+[0..7], kb+16+hf*8+[0..7]
__device__ __forceinline__ v16bf ldfragA(const unsigned short* __restrict__ row,
                                         int kb, int hf) {
    bfpair p;
    p.lo = *(const v8us*)(row + kb + hf * 8);
    p.hi = *(const v8us*)(row + kb + 16 + hf * 8);
    return __builtin_bit_cast(v16bf, p);
}
// B 32x16 bf16 fragment: lane holds column lr; K = kb+hf*8+[0..7], kb+16+hf*8+[0..7]
__device__ __forceinline__ v16bf ldfragB(const unsigned short* __restrict__ row,
                                         int kb, int hf) {
    bfpair p;
    p.lo = *(const v8us*)(row + kb + hf * 8);
    p.hi = *(const v8us*)(row + kb + 16 + hf * 8);
    return __builtin_bit_cast(v16bf, p);
}

// C[M,Nout] = act(A[M,K] @ W[Nout,K]^T + bias); bf16 operands, fp32 accumulate.
// One wave: 16-row x 64-col tile (4 accumulators). K stepped by 64 per iteration
// with ping-pong fragment buffers (no register rotation copies).
// Requires K % 64 == 0 (true for all call sites: 256, 64).
// Output: fp32 to Cf if non-null, else bf16 to Cb.
__global__ __launch_bounds__(256) void gemm_wmma_bf16(
    const unsigned short* __restrict__ A, const unsigned short* __restrict__ W,
    const float* __restrict__ bias, float* __restrict__ Cf,
    unsigned short* __restrict__ Cb, int Mtiles, int K, int ldc, int applyLeaky)
{
    const int lane = threadIdx.x & 31;
    const int wave = threadIdx.x >> 5;
    const int rt = blockIdx.x * 8 + wave;           // wave-uniform
    if (rt >= Mtiles) return;                       // whole-wave exit: EXEC stays full
    const int hf = lane >> 4;
    const int lr = lane & 15;
    const int colBase = blockIdx.y * 64;

    const unsigned short* Arow = A + (size_t)(rt * 16 + lr) * K;
    const unsigned short* Wr[4];
#pragma unroll
    for (int c = 0; c < 4; ++c)
        Wr[c] = W + (size_t)(colBase + c * 16 + lr) * K;

    v8f acc[4];
#pragma unroll
    for (int c = 0; c < 4; ++c)
#pragma unroll
        for (int i = 0; i < 8; ++i) acc[c][i] = 0.0f;

    // prologue: buffer 0 holds fragments for kb = 0
    v16bf a0 = ldfragA(Arow, 0, hf);
    v16bf b0[4];
#pragma unroll
    for (int c = 0; c < 4; ++c) b0[c] = ldfragB(Wr[c], 0, hf);

    for (int kb = 0; kb < K; kb += 64) {
        // buffer 1: fragments for kb+32 (in flight while buffer 0 computes)
        v16bf a1 = ldfragA(Arow, kb + 32, hf);
        v16bf b1[4];
#pragma unroll
        for (int c = 0; c < 4; ++c) b1[c] = ldfragB(Wr[c], kb + 32, hf);
#pragma unroll
        for (int c = 0; c < 4; ++c)
            acc[c] = __builtin_amdgcn_wmma_f32_16x16x32_bf16(
                false, a0, false, b0[c], (short)0, acc[c], false, false);
        if (kb + 64 < K) {                          // refill buffer 0 for kb+64
            __builtin_prefetch(Arow + kb + 128, 0, 3);
            a0 = ldfragA(Arow, kb + 64, hf);
#pragma unroll
            for (int c = 0; c < 4; ++c) b0[c] = ldfragB(Wr[c], kb + 64, hf);
        }
#pragma unroll
        for (int c = 0; c < 4; ++c)
            acc[c] = __builtin_amdgcn_wmma_f32_16x16x32_bf16(
                false, a1, false, b1[c], (short)0, acc[c], false, false);
    }

    const int m0 = rt * 16 + hf * 8;                // C layout: lane half selects M block
#pragma unroll
    for (int c = 0; c < 4; ++c) {
        const int col = colBase + c * 16 + lr;
        const float bv = bias ? bias[col] : 0.0f;
#pragma unroll
        for (int r = 0; r < 8; ++r) {
            float v = acc[c][r] + bv;
            if (applyLeaky) v = (v >= 0.0f) ? v : 0.01f * v;
            if (Cf) Cf[(size_t)(m0 + r) * ldc + col] = v;
            else    Cb[(size_t)(m0 + r) * ldc + col] = f2bf(v);
        }
    }
}

// fp32 -> bf16 streaming conversion, 4 elements/thread.
__global__ void cvt_bf16(const float* __restrict__ in, unsigned short* __restrict__ out,
                         int n4) {
    int i = blockIdx.x * 256 + threadIdx.x;
    if (i < n4) {
        float4 v = ((const float4*)in)[i];
        ushort4 o;
        o.x = f2bf(v.x); o.y = f2bf(v.y); o.z = f2bf(v.z); o.w = f2bf(v.w);
        ((ushort4*)out)[i] = o;
    }
}

// EvolveGCN weight generation: GRU step (h0 = 0) + [64 -> 4096] linear, bf16 out.
__global__ __launch_bounds__(256) void evolve_weight(
    const float* __restrict__ wih, const float* __restrict__ bih,
    const float* __restrict__ bhh, const float* __restrict__ mem_in,
    const float* __restrict__ wt_w, const float* __restrict__ wt_b,
    unsigned short* __restrict__ Wout)
{
    __shared__ float gi[192];
    __shared__ float mem[64];
    const int t = threadIdx.x;
    if (t < 192) {
        float s = bih[t];
        for (int k = 0; k < 64; ++k) s += wih[t * 64 + k] * mem_in[k];
        gi[t] = s;
    }
    __syncthreads();
    if (t < 64) {
        float r = 1.0f / (1.0f + expf(-(gi[t]       + bhh[t])));
        float z = 1.0f / (1.0f + expf(-(gi[64 + t]  + bhh[64 + t])));
        float n = tanhf(gi[128 + t] + r * bhh[128 + t]);
        mem[t] = (1.0f - z) * n;                    // + z*h0, h0 = 0
    }
    __syncthreads();
    for (int j = t; j < 4096; j += 256) {
        float s = wt_b[j];
        for (int k = 0; k < 64; ++k) s += wt_w[j * 64 + k] * mem[k];
        Wout[j] = f2bf(s);
    }
}

__global__ void deg_init(float* __restrict__ deg) {
    int i = blockIdx.x * 256 + threadIdx.x;
    if (i < NNODES) deg[i] = 1.0f;                  // self-loop
}
__global__ void deg_accum(const int* __restrict__ ei, float* __restrict__ deg) {
    int e = blockIdx.x * 256 + threadIdx.x;
    if (e < NEDGES) atomicAdd(&deg[ei[NEDGES + e]], 1.0f);
}
__global__ void deg_inv(const float* __restrict__ deg, float* __restrict__ dinv) {
    int i = blockIdx.x * 256 + threadIdx.x;
    if (i < NNODES) dinv[i] = rsqrtf(deg[i]);       // deg >= 1 always
}

__global__ void gcn_self(const float* __restrict__ dinv, const float* __restrict__ xw,
                         float* __restrict__ agg) {
    int i = blockIdx.x * 256 + threadIdx.x;
    if (i < NNODES * HDIM) { float d = dinv[i >> 6]; agg[i] = d * d * xw[i]; }
}
__global__ void gcn_scatter(const int* __restrict__ ei, const float* __restrict__ dinv,
                            const float* __restrict__ xw, float* __restrict__ agg) {
    unsigned t = blockIdx.x * 256u + threadIdx.x;
    if (t < (unsigned)NEDGES * HDIM) {
        unsigned e = t >> 6; int f = t & 63;
        int r = ei[e], c = ei[NEDGES + e];
        atomicAdd(&agg[(size_t)c * HDIM + f], dinv[r] * dinv[c] * xw[(size_t)r * HDIM + f]);
    }
}
__global__ void gcn_fin_bf16(const float* __restrict__ agg, const float* __restrict__ b,
                             unsigned short* __restrict__ hout) {
    int i = blockIdx.x * 256 + threadIdx.x;
    if (i < NNODES * HDIM) {
        float v = agg[i] + b[i & 63];
        hout[i] = f2bf((v >= 0.0f) ? v : 0.01f * v);
    }
}
__global__ void gcn_fin_f32(const float* __restrict__ agg, const float* __restrict__ b,
                            float* __restrict__ hout) {
    int i = blockIdx.x * 256 + threadIdx.x;
    if (i < NNODES * HDIM) {
        float v = agg[i] + b[i & 63];
        hout[i] = (v >= 0.0f) ? v : 0.01f * v;
    }
}

// One wave per labeled edge: 160-dim dot for score + anomaly heads.
__global__ __launch_bounds__(256) void edge_score(
    const int* __restrict__ eli, const float* __restrict__ ea,
    const float* __restrict__ hhat,
    const float* __restrict__ w1, const float* __restrict__ b1,
    const float* __restrict__ wA, const float* __restrict__ bA,
    float* __restrict__ score, float* __restrict__ anom)
{
    const int wid = (int)((blockIdx.x * 256u + threadIdx.x) >> 5);
    const int lane = threadIdx.x & 31;
    if (wid >= NLBL) return;
    const int s = eli[wid], d = eli[NLBL + wid];
    float a1 = 0.0f, aA = 0.0f;
    for (int j = lane; j < 160; j += 32) {
        float v = (j < 64)  ? hhat[(size_t)s * 64 + j]
                : (j < 128) ? hhat[(size_t)d * 64 + (j - 64)]
                            : ea[(size_t)wid * 32 + (j - 128)];
        a1 += v * w1[j];
        aA += v * wA[j];
    }
#pragma unroll
    for (int o = 16; o > 0; o >>= 1) {
        a1 += __shfl_xor(a1, o, 32);
        aA += __shfl_xor(aA, o, 32);
    }
    if (lane == 0) { score[wid] = a1 + b1[0]; anom[wid] = aA + bA[0]; }
}

extern "C" void kernel_launch(void* const* d_in, const int* in_sizes, int n_in,
                              void* d_out, int out_size, void* d_ws, size_t ws_size,
                              hipStream_t stream) {
    (void)in_sizes; (void)n_in; (void)out_size; (void)ws_size;
    const float* x       = (const float*)d_in[0];
    const int*   ei      = (const int*)  d_in[1];
    const int*   eli     = (const int*)  d_in[2];
    const float* eattr   = (const float*)d_in[3];
    const float* pre1_w  = (const float*)d_in[4];
    const float* pre1_b  = (const float*)d_in[5];
    const float* pre2_w  = (const float*)d_in[6];
    const float* pre2_b  = (const float*)d_in[7];
    const float* c1_wih  = (const float*)d_in[8];
    const float* c1_bih  = (const float*)d_in[10];
    const float* c1_bhh  = (const float*)d_in[11];
    const float* c1_wtw  = (const float*)d_in[12];
    const float* c1_wtb  = (const float*)d_in[13];
    const float* c1_gcnb = (const float*)d_in[14];
    const float* c1_mem  = (const float*)d_in[15];
    const float* c2_wih  = (const float*)d_in[16];
    const float* c2_bih  = (const float*)d_in[18];
    const float* c2_bhh  = (const float*)d_in[19];
    const float* c2_wtw  = (const float*)d_in[20];
    const float* c2_wtb  = (const float*)d_in[21];
    const float* c2_gcnb = (const float*)d_in[22];
    const float* c2_mem  = (const float*)d_in[23];
    const float* p1w     = (const float*)d_in[24];
    const float* p1b     = (const float*)d_in[25];
    const float* pAw     = (const float*)d_in[26];
    const float* pAb     = (const float*)d_in[27];

    // bf16 staging (ushort units, every region 16B aligned)
    unsigned short* us    = (unsigned short*)d_ws;
    unsigned short* xbf   = us;                     // 12,800,000
    unsigned short* h1bf  = xbf   + 12800000;       // 12,800,000
    unsigned short* h2bf  = h1bf  + 12800000;       //  3,200,000
    unsigned short* hc1bf = h2bf  +  3200000;       //  3,200,000
    unsigned short* w1bf  = hc1bf +  3200000;       //     65,536
    unsigned short* w2bf  = w1bf  +    65536;       //     16,384
    unsigned short* wc1bf = w2bf  +    16384;       //      4,096
    unsigned short* wc2bf = wc1bf +     4096;       //      4,096
    // fp32 region starts at 32,090,112 ushorts = 64,180,224 bytes (16B aligned)
    float* fbase = (float*)(us + 32090112);
    float* xw   = fbase;                            // 3,200,000
    float* agg  = xw  + 3200000;                    // 3,200,000
    float* deg  = agg + 3200000;                    //    50,000
    float* dinv = deg +   50000;                    //    50,000

    float* out   = (float*)d_out;
    float* score = out;
    float* anom  = out + NLBL;
    float* hhat  = out + 2 * NLBL;                  // h_hat written straight into d_out

    const dim3 blk(256);
    const int Mtiles = NNODES / 16;                 // 3125
    const int gX = (Mtiles + 7) / 8;                // 391

    // one-time fp32 -> bf16 staging
    cvt_bf16<<<12500, blk, 0, stream>>>(x,      xbf,  3200000);
    cvt_bf16<<<64,    blk, 0, stream>>>(pre1_w, w1bf, 16384);
    cvt_bf16<<<16,    blk, 0, stream>>>(pre2_w, w2bf, 4096);

    // symmetric-norm degrees (shared by both convs)
    deg_init <<<196,  blk, 0, stream>>>(deg);
    deg_accum<<<3125, blk, 0, stream>>>(ei, deg);
    deg_inv  <<<196,  blk, 0, stream>>>(deg, dinv);

    // preprocess MLP (bf16 in, bf16 out)
    gemm_wmma_bf16<<<dim3(gX, 4), blk, 0, stream>>>(xbf,  w1bf, pre1_b, nullptr, h1bf,
                                                    Mtiles, 256, 256, 1);
    gemm_wmma_bf16<<<dim3(gX, 1), blk, 0, stream>>>(h1bf, w2bf, pre2_b, nullptr, h2bf,
                                                    Mtiles, 256, 64, 1);

    // conv1
    evolve_weight<<<1, blk, 0, stream>>>(c1_wih, c1_bih, c1_bhh, c1_mem, c1_wtw, c1_wtb, wc1bf);
    gemm_wmma_bf16<<<dim3(gX, 1), blk, 0, stream>>>(h2bf, wc1bf, nullptr, xw, nullptr,
                                                    Mtiles, 64, 64, 0);
    gcn_self    <<<12500,  blk, 0, stream>>>(dinv, xw, agg);
    gcn_scatter <<<200000, blk, 0, stream>>>(ei, dinv, xw, agg);
    gcn_fin_bf16<<<12500,  blk, 0, stream>>>(agg, c1_gcnb, hc1bf);

    // conv2
    evolve_weight<<<1, blk, 0, stream>>>(c2_wih, c2_bih, c2_bhh, c2_mem, c2_wtw, c2_wtb, wc2bf);
    gemm_wmma_bf16<<<dim3(gX, 1), blk, 0, stream>>>(hc1bf, wc2bf, nullptr, xw, nullptr,
                                                    Mtiles, 64, 64, 0);
    gcn_self   <<<12500,  blk, 0, stream>>>(dinv, xw, agg);
    gcn_scatter<<<200000, blk, 0, stream>>>(ei, dinv, xw, agg);
    gcn_fin_f32<<<12500,  blk, 0, stream>>>(agg, c2_gcnb, hhat);

    // edge scoring heads
    edge_score<<<50000, blk, 0, stream>>>(eli, eattr, hhat, p1w, p1b, pAw, pAb, score, anom);
}
